// Net_66975720014255
// MI455X (gfx1250) — matrix operand order repaired and checked by reading, not compiled, not run on hardware
//
#include <hip/hip_runtime.h>
#include <hip/hip_bf16.h>
#include <math.h>

// ---------------------------------------------------------------------------
// MI455X (gfx1250) implementation.
// Dominant cost: two GCN-hop GEMMs (14080x1600x1600, 2x) = ~144 GFLOP.
// Compute-bound at f32 rates => run on v_wmma_f32_16x16x32_f16 (f16 in,
// f32 accumulate). GEMM uses 64x64 per-wave register tiles (16 WMMA accums)
// and LDS-staged, double-buffered B tiles shared by the 4 waves of a block
// (256x64 block tile) to cut L2 traffic ~3x vs naive per-wave fetch.
// Everything else is memory-bound fused elementwise / thin-GEMM kernels.
// ---------------------------------------------------------------------------

typedef _Float16 h16;
typedef __attribute__((ext_vector_type(16))) _Float16 v16h;
typedef __attribute__((ext_vector_type(8)))  _Float16 v8h;
typedef __attribute__((ext_vector_type(8)))  float    v8f;

union F16x16 { v16h v; v8h h[2]; };

#define BN_SCALE 0.99999500003749968750f  // 1/sqrt(1+1e-5)

// Problem constants
#define BB 32
#define CC 40
#define TT 12
#define NN 800
#define RR 11
#define N2 1600
#define MM (BB*CC*RR)        // 14080

// Workspace layout (bytes, all 256-aligned)
#define OFF_DATA   0u                          // f32  B*C*T*N   = 49,152,000 B
#define OFF_ADJT   49152000u                   // f16  N2*N2     =  5,120,000 B
#define OFF_H      54272000u                   // f16  M*N2      = 45,056,000 B
#define OFF_G      99328000u                   // f32  M*N2      = 90,112,000 B
#define OFF_HFIN   189440000u                  // f32  B*C*R*N   = 45,056,000 B
#define OFF_DRES   234496000u                  // f32  B*C*R*N   = 45,056,000 B

// ---------------------------------------------------------------------------
// 1) data = x + t_emb + s_emb; fused: also emit hop-0 operand h0 in f16.
//    h0[bc*R+t, 0:800] = data[bc,t,:]  (t<=10)
//    h0[bc*R+t-1, 800:1600] = data[bc,t,:]  (t>=1)
// ---------------------------------------------------------------------------
__global__ __launch_bounds__(256)
void k_build_data(const float* __restrict__ x, const float* __restrict__ t_emb,
                  const float* __restrict__ s_emb, float* __restrict__ data,
                  h16* __restrict__ hf) {
    size_t i = (size_t)blockIdx.x * 256 + threadIdx.x;   // B*C*T*N threads
    int n = (int)(i % NN);
    int t = (int)((i / NN) % TT);
    int c = (int)((i / (NN * TT)) % CC);
    size_t bc = i / (NN * TT * (size_t)CC) * CC + c;     // = i / (NN*TT)
    float v = x[i] + t_emb[c * TT + t] + s_emb[c * NN + n];
    data[i] = v;
    h16 vh = (h16)v;
    if (t < RR) hf[((size_t)(bc * RR + t)) * N2 + n] = vh;
    if (t >= 1) hf[((size_t)(bc * RR + t - 1)) * N2 + NN + n] = vh;
}

// ---------------------------------------------------------------------------
// 2) adjT[n][k] = relu( sum_d nv1[k,d] * nv2[d,n] ), stored f16, transposed
// ---------------------------------------------------------------------------
__global__ __launch_bounds__(256)
void k_adjT(const float* __restrict__ nv1, const float* __restrict__ nv2,
            h16* __restrict__ adjT) {
    size_t i = (size_t)blockIdx.x * 256 + threadIdx.x;   // N2*N2 threads
    int k = (int)(i % N2);
    int n = (int)(i / N2);
    float acc = 0.f;
    #pragma unroll
    for (int d = 0; d < 40; ++d)
        acc += nv1[(size_t)k * 40 + d] * nv2[(size_t)d * N2 + n];
    adjT[(size_t)n * N2 + k] = (h16)(acc > 0.f ? acc : 0.f);
}

// ---------------------------------------------------------------------------
// 3) WMMA GEMM: G[m,n] = sum_k hf[m,k] * adjT[n,k]   (f16 x f16 -> f32)
//    Block = 4 waves, block tile 256(M) x 64(N); wave tile 64x64 (4x4 WMMA).
//    B tile (64 cols x 32 K, f16) staged in LDS, double buffered, shared by
//    all 4 waves; 80B LDS row pitch keeps ds_load_b128 bank-conflict-free.
// ---------------------------------------------------------------------------
#define BROW 40                       // LDS B row pitch in halfs (80 B)
__global__ __launch_bounds__(128)
void k_gemm(const h16* __restrict__ A, const h16* __restrict__ Bt,
            float* __restrict__ G) {
    __shared__ h16 sB[2][64 * BROW];  // 2 x 5120 B

    const int tid   = threadIdx.x;
    const int lane  = tid & 31;
    const int wave  = tid >> 5;
    const int bm    = blockIdx.x / 25;           // 0..54   (M/256)
    const int bn    = blockIdx.x - bm * 25;      // 0..24   (N2/64)
    const int m0    = bm * 256 + wave * 64;
    const int n0    = bn * 64;
    const int lhalf = lane >> 4;                 // 0|1
    const int l16   = lane & 15;

    // Cooperative B stage: 128 threads x 16B = 32 rows x 32 halfs
    const int trow = tid >> 2;                   // 0..31
    const int tseg = (tid & 3) * 8;              // 0,8,16,24
    const h16* bsrc = Bt + (size_t)(n0 + trow) * N2 + tseg;
    h16* bdst = &sB[0][0] + trow * BROW + tseg;

    // A fragment base: lane = M row; lanes 0-15 hold K {0..7,16..23},
    // lanes 16-31 hold K {8..15,24..31}
    const h16* aptr = A + (size_t)(m0 + l16) * N2 + lhalf * 8;

    v8f acc[4][4] = {};

    // prologue: stage K-chunk 0 into buffer 0
    *(v8h*)bdst = *(const v8h*)bsrc;
    __syncthreads();

    for (int kb = 0; kb < N2; kb += 32) {
        const int buf = (kb >> 5) & 1;
        // prefetch next B chunk into the other buffer
        if (kb + 32 < N2)
            *(v8h*)(bdst + (buf ^ 1) * (64 * BROW)) = *(const v8h*)(bsrc + kb + 32);

        F16x16 a[4], b[4];
        #pragma unroll
        for (int mi = 0; mi < 4; ++mi) {
            const h16* ap = aptr + (size_t)mi * 16 * N2 + kb;
            a[mi].h[0] = *(const v8h*)(ap);
            a[mi].h[1] = *(const v8h*)(ap + 16);
        }
        #pragma unroll
        for (int ni = 0; ni < 4; ++ni) {
            // B (32x16): lane = N col; lanes 0-15 K {0..7,16..23}, lanes 16-31 K {8..15,24..31}
            const h16* bl = &sB[buf][0] + (ni * 16 + l16) * BROW + lhalf * 8;
            b[ni].h[0] = *(const v8h*)(bl);
            b[ni].h[1] = *(const v8h*)(bl + 16);
        }
        #pragma unroll
        for (int mi = 0; mi < 4; ++mi)
            #pragma unroll
            for (int ni = 0; ni < 4; ++ni)
                acc[mi][ni] = __builtin_amdgcn_wmma_f32_16x16x32_f16(
                    false, a[mi].v, false, b[ni].v, (short)0, acc[mi][ni],
                    false, false);
        __syncthreads();   // all waves done with buf before it is re-staged
    }

    // C/D layout: lane = N col (l16); rows = lhalf*8 + i
    const int rbase = lhalf * 8;
    #pragma unroll
    for (int mi = 0; mi < 4; ++mi) {
        #pragma unroll
        for (int ni = 0; ni < 4; ++ni) {
            float* gp = G + (size_t)(m0 + mi * 16 + rbase) * N2
                          + (n0 + ni * 16 + l16);
            #pragma unroll
            for (int i = 0; i < 8; ++i)
                gp[(size_t)i * N2] = acc[mi][ni][i];
        }
    }
}

// ---------------------------------------------------------------------------
// 4) channel mix + ReLU:  out[b,d,r,n2] = relu( sum_c W[d,c] * G[b,c,r,n2] )
//    LDS-tiled over n2 (tile 64); hop0 -> f16 full row, hop1 -> f32 n2>=800.
// ---------------------------------------------------------------------------
__global__ __launch_bounds__(256)
void k_mix(const float* __restrict__ G, const float* __restrict__ W,
           h16* __restrict__ out_h, float* __restrict__ out_f, int final_hop) {
    __shared__ float sG[CC * 64];
    __shared__ float sW[CC * CC];
    const int tid   = threadIdx.x;
    const int tileN = blockIdx.x % 25;
    const int r     = (blockIdx.x / 25) % RR;
    const int b     = blockIdx.x / (25 * RR);
    const int n2b   = tileN * 64;

    for (int i = tid; i < CC * 64; i += 256) {
        int c = i >> 6, col = i & 63;
        sG[i] = G[((size_t)((b * CC + c) * RR + r)) * N2 + n2b + col];
    }
    for (int i = tid; i < CC * CC; i += 256) sW[i] = W[i];
    __syncthreads();

    for (int o = tid; o < CC * 64; o += 256) {
        int d = o >> 6, col = o & 63;
        float acc = 0.f;
        #pragma unroll
        for (int c = 0; c < CC; ++c)
            acc += sW[d * CC + c] * sG[c * 64 + col];
        acc = acc > 0.f ? acc : 0.f;
        int n2 = n2b + col;
        size_t row = (size_t)((b * CC + d) * RR + r);
        if (final_hop) {
            if (n2 >= NN) out_f[row * NN + (n2 - NN)] = acc;
        } else {
            out_h[row * N2 + n2] = (h16)acc;
        }
    }
}

// ---------------------------------------------------------------------------
// 5) gated temporal conv:  data_res = tanh(f) * sigmoid(g), K=80 over (c,tap)
// ---------------------------------------------------------------------------
__global__ __launch_bounds__(256)
void k_dres(const float* __restrict__ data,
            const float* __restrict__ W_f, const float* __restrict__ b_f,
            const float* __restrict__ W_g, const float* __restrict__ b_g,
            float* __restrict__ dres) {
    __shared__ float sD[CC * 2 * 80];   // [c][tap][col]
    __shared__ float sWf[CC * CC * 2];
    __shared__ float sWg[CC * CC * 2];
    __shared__ float sbf[CC], sbg[CC];
    const int tid   = threadIdx.x;
    const int tileN = blockIdx.x % 10;
    const int r     = (blockIdx.x / 10) % RR;
    const int b     = blockIdx.x / (10 * RR);
    const int nb    = tileN * 80;

    for (int i = tid; i < CC * 2 * 80; i += 256) {
        int c   = i / 160;
        int j   = i - c * 160;
        int tap = j / 80;
        int col = j - tap * 80;
        sD[i] = data[((size_t)(b * CC + c) * TT + (r + tap)) * NN + nb + col];
    }
    for (int i = tid; i < CC * CC * 2; i += 256) { sWf[i] = W_f[i]; sWg[i] = W_g[i]; }
    if (tid < CC) { sbf[tid] = b_f[tid]; sbg[tid] = b_g[tid]; }
    __syncthreads();

    for (int o = tid; o < CC * 80; o += 256) {
        int d = o / 80, col = o - d * 80;
        float f = sbf[d], g = sbg[d];
        #pragma unroll
        for (int c = 0; c < CC; ++c) {
            float x0 = sD[c * 160 + col];
            float x1 = sD[c * 160 + 80 + col];
            f += x0 * sWf[(d * CC + c) * 2 + 0] + x1 * sWf[(d * CC + c) * 2 + 1];
            g += x0 * sWg[(d * CC + c) * 2 + 0] + x1 * sWg[(d * CC + c) * 2 + 1];
        }
        float sig = 1.f / (1.f + expf(-g));
        dres[((size_t)(b * CC + d) * RR + r) * NN + nb + col] = tanhf(f) * sig;
    }
}

// ---------------------------------------------------------------------------
// 6) finalize: out_pre = hfin + dres; skip = (out_pre @ W_skip^T + b)*s;
//    residual = data @ W_res^T + b; out = (out_pre + residual)*s;
//    d_out = concat([out (11), skip (12)], axis=2)  -> (B,C,23,N)
// ---------------------------------------------------------------------------
__global__ __launch_bounds__(256)
void k_finalize(const float* __restrict__ hfin, const float* __restrict__ dres,
                const float* __restrict__ data,
                const float* __restrict__ W_skip, const float* __restrict__ b_skip,
                const float* __restrict__ W_res,  const float* __restrict__ b_res,
                float* __restrict__ out) {
    __shared__ float sWs[12 * RR], sbs[12], sWr[RR * TT], sbr[RR];
    const int tid = threadIdx.x;
    if (tid < 12 * RR) sWs[tid] = W_skip[tid];
    if (tid < 12)      sbs[tid] = b_skip[tid];
    if (tid < RR * TT) sWr[tid] = W_res[tid];
    if (tid < RR)      sbr[tid] = b_res[tid];
    __syncthreads();

    size_t id = (size_t)blockIdx.x * 256 + tid;   // B*C*N threads
    int n = (int)(id % NN);
    size_t bc = id / NN;

    const float* dp = data + bc * (size_t)(TT * NN) + n;
    const float* hp = hfin + bc * (size_t)(RR * NN) + n;
    const float* rp = dres + bc * (size_t)(RR * NN) + n;
    float* op = out + bc * (size_t)(23 * NN) + n;

    float pre[RR], dat[TT];
    #pragma unroll
    for (int r = 0; r < RR; ++r) pre[r] = hp[(size_t)r * NN] + rp[(size_t)r * NN];
    #pragma unroll
    for (int t = 0; t < TT; ++t) dat[t] = dp[(size_t)t * NN];

    #pragma unroll
    for (int s = 0; s < 12; ++s) {
        float acc = sbs[s];
        #pragma unroll
        for (int r = 0; r < RR; ++r) acc += pre[r] * sWs[s * RR + r];
        op[(size_t)(RR + s) * NN] = acc * BN_SCALE;
    }
    #pragma unroll
    for (int r = 0; r < RR; ++r) {
        float acc = sbr[r];
        #pragma unroll
        for (int t = 0; t < TT; ++t) acc += dat[t] * sWr[r * TT + t];
        op[(size_t)r * NN] = (pre[r] + acc) * BN_SCALE;
    }
}

// ---------------------------------------------------------------------------
extern "C" void kernel_launch(void* const* d_in, const int* in_sizes, int n_in,
                              void* d_out, int out_size, void* d_ws, size_t ws_size,
                              hipStream_t stream) {
    const float* x      = (const float*)d_in[0];
    const float* t_emb  = (const float*)d_in[1];
    const float* s_emb  = (const float*)d_in[2];
    const float* nv1    = (const float*)d_in[3];
    const float* nv2    = (const float*)d_in[4];
    const float* W_f    = (const float*)d_in[5];
    const float* b_f    = (const float*)d_in[6];
    const float* W_g    = (const float*)d_in[7];
    const float* b_g    = (const float*)d_in[8];
    const float* W_gcn  = (const float*)d_in[9];
    const float* W_skip = (const float*)d_in[10];
    const float* b_skip = (const float*)d_in[11];
    const float* W_res  = (const float*)d_in[12];
    const float* b_res  = (const float*)d_in[13];
    float* out = (float*)d_out;

    char* ws = (char*)d_ws;
    float* data  = (float*)(ws + OFF_DATA);
    h16*   adjT  = (h16*)  (ws + OFF_ADJT);
    h16*   hf    = (h16*)  (ws + OFF_H);
    float* G     = (float*)(ws + OFF_G);
    float* hfin  = (float*)(ws + OFF_HFIN);
    float* dres  = (float*)(ws + OFF_DRES);

    // 1) data = x + embeddings, fused hop-0 f16 pack   (12,288,000 elems)
    k_build_data<<<48000, 256, 0, stream>>>(x, t_emb, s_emb, data, hf);
    // 2) adjT f16                                      (2,560,000 elems)
    k_adjT<<<10000, 256, 0, stream>>>(nv1, nv2, adjT);
    // 3-4) hop 0: WMMA GEMM + channel mix
    k_gemm<<<1375, 128, 0, stream>>>(hf, adjT, G);      // 55 x 25 block tiles
    k_mix<<<8800, 256, 0, stream>>>(G, W_gcn, hf, nullptr, 0);
    // 3-4) hop 1
    k_gemm<<<1375, 128, 0, stream>>>(hf, adjT, G);
    k_mix<<<8800, 256, 0, stream>>>(G, W_gcn + CC * CC, nullptr, hfin, 1);
    // 5) gated temporal conv
    k_dres<<<3520, 256, 0, stream>>>(data, W_f, b_f, W_g, b_g, dres);
    // 6) fused skip/residual/concat
    k_finalize<<<4000, 256, 0, stream>>>(hfin, dres, data, W_skip, b_skip,
                                         W_res, b_res, out);
}
